// GNN_43370579755357
// MI455X (gfx1250) — matrix operand order repaired and checked by reading, not compiled, not run on hardware
//
#include <hip/hip_runtime.h>
#include <hip/hip_bf16.h>

#define NN 50000
#define TT 4
#define HH 128
#define NHEADS 8
#define DK 16
#define LLAYERS 2
#define EE 1600000
#define NPT 12500

typedef __attribute__((ext_vector_type(16))) __bf16 v16bf;
typedef __attribute__((ext_vector_type(8)))  float  v8f;

__device__ __forceinline__ float gelu_tanh(float x) {
    // jax.nn.gelu default (approximate=True)
    float x3 = x * x * x;
    return 0.5f * x * (1.0f + tanhf(0.7978845608028654f * (x + 0.044715f * x3)));
}

__device__ __forceinline__ void atomicMaxF(float* addr, float val) {
    // sign-split trick: int compare == float compare for >=0; reversed (uint) for <0
    if (val >= 0.0f) atomicMax((int*)addr, __float_as_int(val));
    else             atomicMin((unsigned int*)addr, __float_as_uint(val));
}

// ---------------------------------------------------------------------------
// Pack W[K x 128] (f32, row-major) into WMMA-B-ready bf16:
//   P[((kt*8 + nt)*32 + lane)*16 + 2j {,+1}] = bf16(W[(kt*32 + 16*half + 2j {,+1})*128 + nt*16 + n])
// so each lane's 16 B-operand values are one contiguous 32-byte chunk.
// One thread packs one dword (2 bf16). Works for any K multiple of 32, so the
// stacked [L,T,128,128] tensors pack in one call with K = L*T*128 (block
// boundaries align with k-tiles, leaving per-(l,t) packed blocks contiguous).
// ---------------------------------------------------------------------------
__global__ void hgt_pack_w(const float* __restrict__ W, __bf16* __restrict__ P, int K)
{
    int tid = blockIdx.x * blockDim.x + threadIdx.x;
    int total = (K >> 5) << 11;          // (K/32)*8*32*8 dwords
    if (tid >= total) return;
    int j    = tid & 7;
    int lane = (tid >> 3) & 31;
    int nt   = (tid >> 8) & 7;
    int kt   = tid >> 11;
    int n    = lane & 15;
    int half = lane >> 4;
    int k    = kt * 32 + 16 * half + 2 * j;
    union { __bf16 h[2]; unsigned int u; } pk;
    pk.h[0] = (__bf16)W[(size_t)k * HH + nt * 16 + n];
    pk.h[1] = (__bf16)W[(size_t)(k + 1) * HH + nt * 16 + n];
    ((unsigned int*)P)[tid] = pk.u;
}

// ---------------------------------------------------------------------------
// WMMA GEMM: Y[rows x 128] = A[rows x K] @ W[K x 128] + bias, optional tanh.
// One wave = one 16-row M-tile x all 128 output columns (8 f32 accumulators).
// A: f32 loaded as float2 (row clamped to stay in bounds; tail rows computed
//    but never stored), converted to bf16 in-register (contiguous k order:
//    half 0 holds K 0-15; half 1 holds K 16-31, same as the packed B).
// B: pre-packed bf16 -> one 32-byte contiguous load per lane per WMMA; all 8
//    N-tiles loaded up front so B fetch overlaps the WMMA chain.
// ---------------------------------------------------------------------------
__global__ void hgt_gemm_wmma(const float* __restrict__ A,
                              const __bf16* __restrict__ Wp,
                              const float* __restrict__ bias,
                              float* __restrict__ Y,
                              int rows, int K, int tanh_out)
{
    const int lane = threadIdx.x & 31;
    const int wave = threadIdx.x >> 5;
    const int mtile = blockIdx.x * 8 + wave;
    const int row0 = mtile * 16;
    if (row0 >= rows) return;                 // wave-uniform exit (EXEC all-1 at WMMA)

    const int m    = lane & 15;
    const int half = lane >> 4;
    const int n    = lane & 15;
    // clamp instead of predicate: branch-free loads; clamped rows' outputs are
    // discarded by the store guard below.
    const int arow = (row0 + m < rows) ? (row0 + m) : (rows - 1);
    const float2* Arow2 = (const float2*)(A + (size_t)arow * K);
    const v16bf* Bp = (const v16bf*)Wp;

    v8f acc[8] = {};
    const int ktiles = K >> 5;

    for (int kt = 0; kt < ktiles; ++kt) {
        v16bf a;
        #pragma unroll
        for (int j = 0; j < 8; ++j) {
            int idx2 = (kt << 4) + j + 8 * half;
            float2 f = Arow2[idx2];
            a[2 * j]     = (__bf16)f.x;
            a[2 * j + 1] = (__bf16)f.y;
        }
        v16bf btile[8];
        #pragma unroll
        for (int nt = 0; nt < 8; ++nt)
            btile[nt] = Bp[(size_t)((kt * 8 + nt) * 32 + lane)];
        #pragma unroll
        for (int nt = 0; nt < 8; ++nt)
            acc[nt] = __builtin_amdgcn_wmma_f32_16x16x32_bf16(
                false, a, false, btile[nt], (short)0, acc[nt], false, false);
    }

    // C/D: lane holds column n, VGPR r = row r + 8*half
    #pragma unroll
    for (int nt = 0; nt < 8; ++nt) {
        #pragma unroll
        for (int r = 0; r < 8; ++r) {
            int row = row0 + r + 8 * half;
            if (row < rows) {
                int col = nt * 16 + n;
                float y = acc[nt][r] + bias[col];
                if (tanh_out) y = tanhf(y);
                Y[(size_t)row * HH + col] = y;
            }
        }
    }
}

// k_hat[n,h,f] = (1/sqrt(DK)) * sum_d k[n,h,d] * h_mat[h,d,f]
__global__ void hgt_ktrans(const float* __restrict__ k,
                           const float* __restrict__ hmat,
                           float* __restrict__ kt)
{
    int tid = blockIdx.x * blockDim.x + threadIdx.x;
    if (tid >= NN * HH) return;
    int node = tid >> 7, c = tid & 127, h = c >> 4, f = c & 15;
    const float* kr = k + (size_t)node * HH + h * DK;
    const float* hm = hmat + h * DK * DK + f;
    float s = 0.0f;
    #pragma unroll
    for (int d = 0; d < DK; ++d) s += kr[d] * hm[d * DK];
    kt[tid] = 0.25f * s;   // fold 1/sqrt(16)
}

__global__ void hgt_init(float* __restrict__ mbuf, float* __restrict__ den,
                         float* __restrict__ agg)
{
    int tid = blockIdx.x * blockDim.x + threadIdx.x;
    if (tid < NN * HH) agg[tid] = 0.0f;
    if (tid < NN * NHEADS) { mbuf[tid] = -__builtin_inff(); den[tid] = 0.0f; }
}

// pass A: att[e,h] = dot16(k_hat[src,h,:], q[dst,h,:]); running segment max
__global__ void hgt_edge_logits(const int* __restrict__ ei,
                                const float* __restrict__ kt,
                                const float* __restrict__ q,
                                float* __restrict__ att,
                                float* __restrict__ mbuf)
{
    long long tid = (long long)blockIdx.x * blockDim.x + threadIdx.x;
    if (tid >= (long long)EE * NHEADS) return;
    int e = (int)(tid >> 3), h = (int)(tid & 7);
    int src = ei[e], dst = ei[EE + e];
    const float4* pk = (const float4*)(kt + (size_t)src * HH + h * DK);
    const float4* pq = (const float4*)(q  + (size_t)dst * HH + h * DK);
    float s = 0.0f;
    #pragma unroll
    for (int i = 0; i < 4; ++i) {
        float4 ka = pk[i], qa = pq[i];
        s += ka.x * qa.x + ka.y * qa.y + ka.z * qa.z + ka.w * qa.w;
    }
    att[tid] = s;
    atomicMaxF(&mbuf[(size_t)dst * NHEADS + h], s);
}

// pass B: ex = exp(att - max); den += ex; agg[dst] += v[src] * ex
// (alpha = ex/den is applied once per node afterwards)
__global__ void hgt_edge_accum(const int* __restrict__ ei,
                               const float* __restrict__ v,
                               const float* __restrict__ att,
                               const float* __restrict__ mbuf,
                               float* __restrict__ den,
                               float* __restrict__ agg)
{
    long long tid = (long long)blockIdx.x * blockDim.x + threadIdx.x;
    if (tid >= (long long)EE * NHEADS) return;
    int e = (int)(tid >> 3), h = (int)(tid & 7);
    int src = ei[e], dst = ei[EE + e];
    float ex = expf(att[tid] - mbuf[(size_t)dst * NHEADS + h]);
    atomicAdd(&den[(size_t)dst * NHEADS + h], ex);
    const float4* pv = (const float4*)(v + (size_t)src * HH + h * DK);
    float* pa = agg + (size_t)dst * HH + h * DK;
    #pragma unroll
    for (int i = 0; i < 4; ++i) {
        float4 vv = pv[i];
        atomicAdd(pa + 4 * i + 0, vv.x * ex);
        atomicAdd(pa + 4 * i + 1, vv.y * ex);
        atomicAdd(pa + 4 * i + 2, vv.z * ex);
        atomicAdd(pa + 4 * i + 3, vv.w * ex);
    }
}

// agg = gelu(agg / (den + 1e-16))
__global__ void hgt_softmax_gelu(float* __restrict__ agg, const float* __restrict__ den)
{
    int tid = blockIdx.x * blockDim.x + threadIdx.x;
    if (tid >= NN * HH) return;
    int node = tid >> 7, h = (tid & 127) >> 4;
    float val = agg[tid] / (den[(size_t)node * NHEADS + h] + 1e-16f);
    agg[tid] = gelu_tanh(val);
}

// out = LN(o*a + x*(1-a)) * g + b ; one wave per node (4 cols/lane)
__global__ void hgt_skip_ln(const float* __restrict__ o,
                            const float* __restrict__ x,
                            const float* __restrict__ skip,   // skip + l*T
                            const float* __restrict__ g,
                            const float* __restrict__ b,
                            float* __restrict__ out)
{
    int lane = threadIdx.x & 31;
    int wave = threadIdx.x >> 5;
    int node = blockIdx.x * 8 + wave;
    if (node >= NN) return;
    int t = node / NPT;
    float a = 1.0f / (1.0f + expf(-skip[t]));

    float vals[4];
    float s = 0.0f;
    #pragma unroll
    for (int i = 0; i < 4; ++i) {
        int c = lane + 32 * i;
        size_t idx = (size_t)node * HH + c;
        float mix = o[idx] * a + x[idx] * (1.0f - a);
        vals[i] = mix;
        s += mix;
    }
    #pragma unroll
    for (int off = 16; off > 0; off >>= 1) s += __shfl_xor(s, off, 32);
    float mean = s * (1.0f / HH);
    float ss = 0.0f;
    #pragma unroll
    for (int i = 0; i < 4; ++i) { float d = vals[i] - mean; ss += d * d; }
    #pragma unroll
    for (int off = 16; off > 0; off >>= 1) ss += __shfl_xor(ss, off, 32);
    float inv = rsqrtf(ss * (1.0f / HH) + 1e-5f);
    #pragma unroll
    for (int i = 0; i < 4; ++i) {
        int c = lane + 32 * i;
        out[(size_t)node * HH + c] = (vals[i] - mean) * inv * g[c] + b[c];
    }
}

extern "C" void kernel_launch(void* const* d_in, const int* in_sizes, int n_in,
                              void* d_out, int out_size, void* d_ws, size_t ws_size,
                              hipStream_t stream)
{
    const float* F[4]  = {(const float*)d_in[0], (const float*)d_in[1],
                          (const float*)d_in[2], (const float*)d_in[3]};
    const int*   ei    = (const int*)d_in[5];
    const float* hmat  = (const float*)d_in[6];
    const float* aw[4] = {(const float*)d_in[7], (const float*)d_in[9],
                          (const float*)d_in[11], (const float*)d_in[13]};
    const float* ab[4] = {(const float*)d_in[8], (const float*)d_in[10],
                          (const float*)d_in[12], (const float*)d_in[14]};
    const float* Wk = (const float*)d_in[15];
    const float* Wq = (const float*)d_in[16];
    const float* Wv = (const float*)d_in[17];
    const float* Wo = (const float*)d_in[18];
    const float* bk = (const float*)d_in[19];
    const float* bq = (const float*)d_in[20];
    const float* bv = (const float*)d_in[21];
    const float* bo = (const float*)d_in[22];
    const float* skip = (const float*)d_in[23];
    const float* lng  = (const float*)d_in[24];
    const float* lnb  = (const float*)d_in[25];

    const size_t NH = (size_t)NN * HH;
    float* ws   = (float*)d_ws;
    float* x    = ws;
    float* kb   = x   + NH;
    float* qb   = kb  + NH;
    float* vb   = qb  + NH;
    float* ktb  = vb  + NH;
    float* agg  = ktb + NH;
    float* att  = agg + NH;
    float* mb   = att + (size_t)EE * NHEADS;
    float* den  = mb  + (size_t)NN * NHEADS;
    float* otmp = kb;          // k no longer needed once ktb is built

    // packed bf16 weight regions (32B-aligned: all float offsets are /32)
    const size_t WBLK = (size_t)LLAYERS * TT * HH * HH;   // elems per stacked tensor
    __bf16* pwk = (__bf16*)(den + (size_t)NN * NHEADS);
    __bf16* pwq = pwk + WBLK;
    __bf16* pwv = pwq + WBLK;
    __bf16* pwo = pwv + WBLK;
    const int IND[4] = {256, 128, 512, 64};
    __bf16* pwa[4];
    {
        __bf16* p = pwo + WBLK;
        for (int t = 0; t < TT; ++t) { pwa[t] = p; p += (size_t)IND[t] * HH; }
    }

    dim3 blk(256);
    const int mtiles  = (NPT + 15) / 16;      // 782
    const int gblocks = (mtiles + 7) / 8;     // 8 waves/block
    const int ngrid   = (NN * HH + 255) / 256;
    const long long ethreads = (long long)EE * NHEADS;
    const int egrid  = (int)((ethreads + 255) / 256);

    // ---- pack all weights once (~650K elements, negligible) ----
    {
        int Kw = LLAYERS * TT * HH;                       // 1024 stacked rows
        int tot = (Kw >> 5) << 11;
        int g = (tot + 255) / 256;
        hgt_pack_w<<<g, blk, 0, stream>>>(Wk, pwk, Kw);
        hgt_pack_w<<<g, blk, 0, stream>>>(Wq, pwq, Kw);
        hgt_pack_w<<<g, blk, 0, stream>>>(Wv, pwv, Kw);
        hgt_pack_w<<<g, blk, 0, stream>>>(Wo, pwo, Kw);
        for (int t = 0; t < TT; ++t) {
            int tota = (IND[t] >> 5) << 11;
            hgt_pack_w<<<(tota + 255) / 256, blk, 0, stream>>>(aw[t], pwa[t], IND[t]);
        }
    }

    // input adapters: x_t = tanh(F_t @ aw_t + ab_t) (block-sorted concat)
    for (int t = 0; t < TT; ++t)
        hgt_gemm_wmma<<<gblocks, blk, 0, stream>>>(
            F[t], pwa[t], ab[t], x + (size_t)t * NPT * HH, NPT, IND[t], 1);

    for (int l = 0; l < LLAYERS; ++l) {
        for (int t = 0; t < TT; ++t) {
            size_t woff = ((size_t)l * TT + t) * HH * HH;   // same for f32 and packed
            size_t boff = ((size_t)l * TT + t) * HH;
            size_t xoff = (size_t)t * NPT * HH;
            hgt_gemm_wmma<<<gblocks, blk, 0, stream>>>(x + xoff, pwk + woff, bk + boff, kb + xoff, NPT, HH, 0);
            hgt_gemm_wmma<<<gblocks, blk, 0, stream>>>(x + xoff, pwq + woff, bq + boff, qb + xoff, NPT, HH, 0);
            hgt_gemm_wmma<<<gblocks, blk, 0, stream>>>(x + xoff, pwv + woff, bv + boff, vb + xoff, NPT, HH, 0);
        }
        hgt_ktrans<<<ngrid, blk, 0, stream>>>(kb, hmat, ktb);
        hgt_init<<<ngrid, blk, 0, stream>>>(mb, den, agg);
        hgt_edge_logits<<<egrid, blk, 0, stream>>>(ei, ktb, qb, att, mb);
        hgt_edge_accum <<<egrid, blk, 0, stream>>>(ei, vb, att, mb, den, agg);
        hgt_softmax_gelu<<<ngrid, blk, 0, stream>>>(agg, den);
        for (int t = 0; t < TT; ++t) {
            size_t woff = ((size_t)l * TT + t) * HH * HH;
            size_t boff = ((size_t)l * TT + t) * HH;
            size_t xoff = (size_t)t * NPT * HH;
            hgt_gemm_wmma<<<gblocks, blk, 0, stream>>>(agg + xoff, pwo + woff, bo + boff, otmp + xoff, NPT, HH, 0);
        }
        float* xout = (l == LLAYERS - 1) ? (float*)d_out : x;
        hgt_skip_ln<<<(NN + 7) / 8, blk, 0, stream>>>(
            otmp, x, skip + l * TT, lng + l * HH, lnb + l * HH, xout);
    }
    (void)in_sizes; (void)n_in; (void)out_size; (void)ws_size;
}
